// EdgeGTEncoderLayer_7576322310509
// MI455X (gfx1250) — matrix operand order repaired and checked by reading, not compiled, not run on hardware
//
#include <hip/hip_runtime.h>
#include <math.h>

#define NB  4
#define NN  512
#define ND  64
#define NH  8
#define NHD 8

typedef __attribute__((ext_vector_type(16))) _Float16 v16h;
typedef __attribute__((ext_vector_type(8)))  _Float16 v8h;
typedef __attribute__((ext_vector_type(8)))  float    v8f;

// async-load builtin operand types: 4 x i32 vector in AS1 (global) / AS3 (LDS)
typedef int v4i __attribute__((vector_size(16)));
typedef __attribute__((address_space(1))) v4i gv4i;
typedef __attribute__((address_space(3))) v4i lv4i;

// CDNA5 async global->LDS copy (tracked by ASYNCcnt).
// generic->AS1: numeric value identical; generic->AS3: low 32 bits of a
// generic shared pointer are the LDS byte offset (aperture rule, ISA 10.2).
static __device__ __forceinline__ void async_load_b128(const void* g, void* l) {
#if __has_builtin(__builtin_amdgcn_global_load_async_to_lds_b128)
  __builtin_amdgcn_global_load_async_to_lds_b128(
      (gv4i*)(unsigned long long)g,
      (lv4i*)(unsigned)(unsigned long long)l, 0, 0);
#else
  unsigned loff = (unsigned)(unsigned long long)l;
  asm volatile("global_load_async_to_lds_b128 %0, %1, off"
               :: "v"(loff), "v"((unsigned long long)g) : "memory");
#endif
}

template <int N>
static __device__ __forceinline__ void wait_asynccnt() {
#if __has_builtin(__builtin_amdgcn_s_wait_asynccnt)
  __builtin_amdgcn_s_wait_asynccnt(N);
#else
  asm volatile("s_wait_asynccnt %0" :: "i"(N) : "memory");
#endif
}

// A-matrix (f16 16x32) K index for VGPR v, half hh, lane-group g (ISA 7.12.2)
static __device__ __forceinline__ int a_kidx(int v, int hh, int g) {
  int base = (v < 4) ? (2 * v) : (16 + 2 * (v - 4));
  return base + 8 * g + hh;
}
// B-matrix (f16 32x16) K index: same K order as A (per VGPR v, half hh, lane-group g)
static __device__ __forceinline__ int b_kidx(int v, int hh, int g) {
  return a_kidx(v, hh, g);
}

static __device__ __forceinline__ v8f wmma_f16(v16h a, v16h b, v8f c) {
  return __builtin_amdgcn_wmma_f32_16x16x32_f16(false, a, false, b, (short)0, c,
                                                false, false);
}

// ---------------------------------------------------------------------------
// Kernel 1: ln1(h) -> hn, qkv = hn @ W_h via WMMA, scatter to q/k/v f16 bufs
// grid 128 blocks x 128 threads; block = 16 rows of h
// ---------------------------------------------------------------------------
__global__ __launch_bounds__(128) void k_ln_qkv(
    const float* __restrict__ h, const float* __restrict__ W_h,
    const float* __restrict__ g1, const float* __restrict__ b1,
    _Float16* __restrict__ qf, _Float16* __restrict__ kf,
    _Float16* __restrict__ vf) {
  __shared__ float    rowb[16][ND];
  __shared__ _Float16 hn[16][ND];
  const int tid  = threadIdx.x;
  const int row0 = blockIdx.x * 16;

  for (int idx = tid; idx < 16 * ND; idx += 128)
    rowb[idx >> 6][idx & 63] = h[row0 * ND + idx];
  __syncthreads();

  if (tid < 16) {
    float mu = 0.f;
    for (int d = 0; d < ND; ++d) mu += rowb[tid][d];
    mu *= (1.f / ND);
    float var = 0.f;
    for (int d = 0; d < ND; ++d) { float x = rowb[tid][d] - mu; var += x * x; }
    var *= (1.f / ND);
    float rs = rsqrtf(var + 1e-5f);
    for (int d = 0; d < ND; ++d)
      hn[tid][d] = (_Float16)(((rowb[tid][d] - mu) * rs) * g1[d] + b1[d]);
  }
  __syncthreads();

  const int wave = tid >> 5, lane = tid & 31;
  const int g = lane >> 4, nl = lane & 15;

  for (int t = wave * 3; t < wave * 3 + 3; ++t) {  // 12 col tiles of 16
    v8f c = {};
#pragma unroll
    for (int ks = 0; ks < 2; ++ks) {
      v16h a, bf;
#pragma unroll
      for (int v = 0; v < 8; ++v)
#pragma unroll
        for (int hh = 0; hh < 2; ++hh) {
          a[2 * v + hh]  = hn[nl][ks * 32 + a_kidx(v, hh, g)];
          int kk         = ks * 32 + b_kidx(v, hh, g);
          bf[2 * v + hh] = (_Float16)W_h[kk * (3 * ND) + t * 16 + nl];
        }
      c = wmma_f16(a, bf, c);
    }
#pragma unroll
    for (int r = 0; r < 8; ++r) {
      int grow = row0 + r + 8 * g;
      int b = grow >> 9, n = grow & (NN - 1);
      int col = t * 16 + nl;
      int which = col >> 6, c64 = col & 63;
      int head = c64 >> 3, d = c64 & 7;
      float val = c[r];
      if (which == 0) val *= 0.35355339059327373f;  // 1/sqrt(HD)
      _Float16* dst = (which == 0) ? qf : (which == 1) ? kf : vf;
      dst[(((b * NH + head) * NN) + n) * NHD + d] = (_Float16)val;
    }
  }
}

// ---------------------------------------------------------------------------
// Kernel 2: fused  ew = e @ W_e  (WMMA)  +  flash attention with e1 bias and
// post-softmax gate e2.  grid (32 i-tiles, 4 batches) x 256 threads.
// wave w == head w.  Streams e exactly once (268 MB total) with CDNA5 async
// global->LDS DMA, double-buffered so the next 64 KB tile lands while the
// current one is consumed by WMMA + online softmax.
// ---------------------------------------------------------------------------
__global__ __launch_bounds__(256) void k_edge_attn(
    const float* __restrict__ e, const float* __restrict__ W_e,
    const _Float16* __restrict__ qf, const _Float16* __restrict__ kf,
    const _Float16* __restrict__ vf, float* __restrict__ ybuf) {
  __shared__ float    elds[2][256 * ND];    // 2 x 64 KB (f32 tiles, async DMA)
  __shared__ float    ewlds[256 * 16];      // 16 KB (rows x 2H)
  __shared__ _Float16 qlds[NH * 16 * NHD];  // 2 KB

  const int tid = threadIdx.x;
  const int it  = blockIdx.x;       // q tile
  const int b   = blockIdx.y;
  const int i0  = it * 16;
  const int wave = tid >> 5, lane = tid & 31;
  const int g = lane >> 4, jl = lane & 15;
  const int hh = wave;              // head

  // preload q tile for all heads into LDS
  for (int idx = tid; idx < NH * 16 * NHD; idx += 256) {
    int hd = idx >> 7, rem = idx & 127, i = rem >> 3, d = rem & 7;
    qlds[idx] = qf[(((b * NH + hd) * NN) + (i0 + i)) * NHD + d];
  }
  // preload W_e B-fragments (K=64 -> two 32-row k-steps)
  v16h bw[2];
#pragma unroll
  for (int ks = 0; ks < 2; ++ks)
#pragma unroll
    for (int v = 0; v < 8; ++v)
#pragma unroll
      for (int h2 = 0; h2 < 2; ++h2)
        bw[ks][2 * v + h2] =
            (_Float16)W_e[(ks * 32 + b_kidx(v, h2, g)) * (2 * NH) + jl];

  // issue async DMA for tile 0 (16 x b128 per thread = 64 KB per block)
  {
#pragma unroll
    for (int k = 0; k < 16; ++k) {
      int i4 = tid + k * 256;
      int ch = i4 >> 8, rem = i4 & 255;  // chunk ch = one i row (16j x 64d)
      const float* gp =
          e + ((size_t)((b * NN + i0 + ch) * NN + 0)) * ND + rem * 4;
      async_load_b128(gp, &elds[0][i4 * 4]);
    }
  }
  __syncthreads();

  // hoist this lane's q rows (rows r+8g, head hh)
  float qreg[8][8];
#pragma unroll
  for (int r = 0; r < 8; ++r) {
    int il = r + 8 * g;
#pragma unroll
    for (int d = 0; d < 8; ++d)
      qreg[r][d] = (float)qlds[(hh * 16 + il) * NHD + d];
  }

  float m[8], l[8], acc[8][8];
#pragma unroll
  for (int r = 0; r < 8; ++r) {
    m[r] = -3.0e38f; l[r] = 0.f;
#pragma unroll
    for (int d = 0; d < 8; ++d) acc[r][d] = 0.f;
  }

  for (int jt = 0; jt < NN / 16; ++jt) {
    const int cur = jt & 1;
    // ---- kick off DMA for the NEXT tile into the other buffer ----
    if (jt + 1 < NN / 16) {
#pragma unroll
      for (int k = 0; k < 16; ++k) {
        int i4 = tid + k * 256;
        int ch = i4 >> 8, rem = i4 & 255;
        const float* gp =
            e + ((size_t)((b * NN + i0 + ch) * NN + (jt + 1) * 16)) * ND +
            rem * 4;
        async_load_b128(gp, &elds[cur ^ 1][i4 * 4]);
      }
      // async loads complete in order: <=16 outstanding means tile jt landed
      wait_asynccnt<16>();
    } else {
      wait_asynccnt<0>();
    }
    __syncthreads();  // all waves' portions of tile jt are in LDS

    const float* ebuf = elds[cur];

    // ---- ew = e_tile @ W_e : 16 M-tiles over 8 waves (2 each), K=64 ----
#pragma unroll
    for (int mt = 0; mt < 2; ++mt) {
      const int tile = wave * 2 + mt;
      const int trow = tile * 16 + (lane & 15);
      v8f c = {};
#pragma unroll
      for (int ks = 0; ks < 2; ++ks) {
        v16h a;
#pragma unroll
        for (int v = 0; v < 8; ++v)
#pragma unroll
          for (int h2 = 0; h2 < 2; ++h2)
            a[2 * v + h2] =
                (_Float16)ebuf[trow * ND + ks * 32 + a_kidx(v, h2, g)];
        c = wmma_f16(a, bw[ks], c);
      }
#pragma unroll
      for (int r = 0; r < 8; ++r)
        ewlds[(tile * 16 + r + 8 * g) * 16 + jl] = c[r];
    }
    __syncthreads();

    // ---- flash attention step for this head, key j = jt*16 + jl ----
    const int j = jt * 16 + jl;
    v8h kvh = *(const v8h*)(kf + (((b * NH + hh) * NN) + j) * NHD);
    v8h vvh = *(const v8h*)(vf + (((b * NH + hh) * NN) + j) * NHD);
    float kv[8], vv[8];
#pragma unroll
    for (int d = 0; d < 8; ++d) { kv[d] = (float)kvh[d]; vv[d] = (float)vvh[d]; }

#pragma unroll
    for (int r = 0; r < 8; ++r) {
      const int il = r + 8 * g;
      float s = ewlds[(il * 16 + jl) * 16 + hh];        // e1 bias
#pragma unroll
      for (int d = 0; d < 8; ++d) s = fmaf(qreg[r][d], kv[d], s);
      float e2 = ewlds[(il * 16 + jl) * 16 + NH + hh];  // gate
      float smax = s;
#pragma unroll
      for (int off = 1; off < 16; off <<= 1)
        smax = fmaxf(smax, __shfl_xor(smax, off, 16));
      float mn = fmaxf(m[r], smax);
      float sc = __expf(m[r] - mn);
      float p  = __expf(s - mn);
      l[r] = l[r] * sc + p;
      float pg = p * e2;
#pragma unroll
      for (int d = 0; d < 8; ++d) acc[r][d] = acc[r][d] * sc + pg * vv[d];
      m[r] = mn;
    }
    __syncthreads();  // compute done; next iter may DMA into the other buffer
  }

  // ---- reduce across the 16 key-lanes, normalize, write y ----
#pragma unroll
  for (int r = 0; r < 8; ++r) {
    float lt = l[r];
#pragma unroll
    for (int off = 1; off < 16; off <<= 1) lt += __shfl_xor(lt, off, 16);
    float inv = 1.0f / lt;
#pragma unroll
    for (int d = 0; d < 8; ++d) {
      float a = acc[r][d];
#pragma unroll
      for (int off = 1; off < 16; off <<= 1) a += __shfl_xor(a, off, 16);
      if (jl == d)
        ybuf[((size_t)(b * NN) + i0 + r + 8 * g) * ND + hh * NHD + d] = a * inv;
    }
  }
}

// ---------------------------------------------------------------------------
// Kernel 3: z = ln2(y + h); out = relu(z@w1)@w2 + y   (both GEMMs via WMMA)
// grid 128 blocks x 128 threads; block = 16 rows
// ---------------------------------------------------------------------------
__global__ __launch_bounds__(128) void k_mlp(
    const float* __restrict__ h, const float* __restrict__ ybuf,
    const float* __restrict__ w1, const float* __restrict__ w2,
    const float* __restrict__ g2, const float* __restrict__ b2,
    float* __restrict__ out) {
  __shared__ float    ylds[16][ND];       // 4 KB
  __shared__ _Float16 zlds[16][ND];       // 2 KB
  __shared__ _Float16 t1[16][4 * ND];     // 8 KB
  const int tid  = threadIdx.x;
  const int row0 = blockIdx.x * 16;

  for (int idx = tid; idx < 16 * ND; idx += 128)
    ylds[idx >> 6][idx & 63] = ybuf[row0 * ND + idx];
  __syncthreads();

  if (tid < 16) {
    float u[ND];
    float mu = 0.f;
    for (int d = 0; d < ND; ++d) {
      u[d] = ylds[tid][d] + h[(row0 + tid) * ND + d];
      mu += u[d];
    }
    mu *= (1.f / ND);
    float var = 0.f;
    for (int d = 0; d < ND; ++d) { float x = u[d] - mu; var += x * x; }
    var *= (1.f / ND);
    float rs = rsqrtf(var + 1e-5f);
    for (int d = 0; d < ND; ++d)
      zlds[tid][d] = (_Float16)(((u[d] - mu) * rs) * g2[d] + b2[d]);
  }
  __syncthreads();

  const int wave = tid >> 5, lane = tid & 31;
  const int g = lane >> 4, nl = lane & 15;

  // GEMM1: [16,64] @ [64,256], relu -> t1 (f16)
  for (int t = wave * 4; t < wave * 4 + 4; ++t) {  // 16 tiles over 4 waves
    v8f c = {};
#pragma unroll
    for (int ks = 0; ks < 2; ++ks) {
      v16h a, bf;
#pragma unroll
      for (int v = 0; v < 8; ++v)
#pragma unroll
        for (int h2 = 0; h2 < 2; ++h2) {
          a[2 * v + h2]  = zlds[nl][ks * 32 + a_kidx(v, h2, g)];
          int kk         = ks * 32 + b_kidx(v, h2, g);
          bf[2 * v + h2] = (_Float16)w1[kk * (4 * ND) + t * 16 + nl];
        }
      c = wmma_f16(a, bf, c);
    }
#pragma unroll
    for (int r = 0; r < 8; ++r) {
      float v = c[r];
      t1[r + 8 * g][t * 16 + nl] = (_Float16)(v > 0.f ? v : 0.f);
    }
  }
  __syncthreads();

  // GEMM2: [16,256] @ [256,64], add y, write out
  const int t = wave;  // 4 col tiles over 4 waves
  v8f c = {};
#pragma unroll
  for (int ks = 0; ks < 8; ++ks) {
    v16h a, bf;
#pragma unroll
    for (int v = 0; v < 8; ++v)
#pragma unroll
      for (int h2 = 0; h2 < 2; ++h2) {
        a[2 * v + h2]  = t1[nl][ks * 32 + a_kidx(v, h2, g)];
        int kk         = ks * 32 + b_kidx(v, h2, g);
        bf[2 * v + h2] = (_Float16)w2[kk * ND + t * 16 + nl];
      }
    c = wmma_f16(a, bf, c);
  }
#pragma unroll
  for (int r = 0; r < 8; ++r) {
    int row = r + 8 * g;
    out[(row0 + row) * ND + t * 16 + nl] = c[r] + ylds[row][t * 16 + nl];
  }
}

// ---------------------------------------------------------------------------
extern "C" void kernel_launch(void* const* d_in, const int* in_sizes, int n_in,
                              void* d_out, int out_size, void* d_ws,
                              size_t ws_size, hipStream_t stream) {
  (void)in_sizes; (void)n_in; (void)out_size; (void)ws_size;
  const float* h   = (const float*)d_in[0];
  const float* e   = (const float*)d_in[1];
  const float* W_h = (const float*)d_in[2];
  const float* W_e = (const float*)d_in[3];
  const float* w1  = (const float*)d_in[4];
  const float* w2  = (const float*)d_in[5];
  const float* g1  = (const float*)d_in[6];
  const float* b1  = (const float*)d_in[7];
  const float* g2  = (const float*)d_in[8];
  const float* b2  = (const float*)d_in[9];
  float* out = (float*)d_out;

  char* ws = (char*)d_ws;
  const size_t QKV_BYTES = (size_t)NB * NH * NN * NHD * sizeof(_Float16);  // 256 KB
  _Float16* qf = (_Float16*)(ws);
  _Float16* kf = (_Float16*)(ws + QKV_BYTES);
  _Float16* vf = (_Float16*)(ws + 2 * QKV_BYTES);
  float*    yb = (float*)(ws + 3 * QKV_BYTES);  // 512 KB

  k_ln_qkv<<<dim3(NB * NN / 16), dim3(128), 0, stream>>>(h, W_h, g1, b1, qf, kf, vf);
  k_edge_attn<<<dim3(NN / 16, NB), dim3(256), 0, stream>>>(e, W_e, qf, kf, vf, yb);
  k_mlp<<<dim3(NB * NN / 16), dim3(128), 0, stream>>>(h, yb, w1, w2, g2, b2, out);
}
